// finetune_ib_73023033967411
// MI455X (gfx1250) — compile-verified, matrix operand order checked by reading, not run on hardware
//
#include <hip/hip_runtime.h>
#include <hip/hip_bf16.h>

typedef __attribute__((ext_vector_type(16))) _Float16 v16h;
typedef __attribute__((ext_vector_type(8)))  _Float16 v8h;
typedef __attribute__((ext_vector_type(8)))  float    v8f;

#define EMB     128
#define C2      256         // 2*EMB
#define KDIM    256         // K dimension of both dense GEMMs
#define NLAYER  5
#define BN_EPS  1e-5f
#define BN_BLOCKS 512

// ---------------------------------------------------------------------------
// Precompute A9[i][0..8] = sum of edge_attr over edges with row==i  (+self loop
// one-hot at slot 7), deg[i] = #edges with row==i (+1 self loop), as float.
// segment_sum(ea @ W + b) == segment_sum(ea) @ W + deg*b, and segment_sum(ea)
// is layer-invariant -> the 850k-edge edge-MLP collapses to [N,9]@[9,128].
// ---------------------------------------------------------------------------
__global__ __launch_bounds__(256)
void k_a9_init(float* __restrict__ A9, float* __restrict__ deg, int n) {
    int i = blockIdx.x * 256 + threadIdx.x;
    if (i >= n) return;
    float* a = A9 + (size_t)i * 9;
#pragma unroll
    for (int k = 0; k < 9; ++k) a[k] = (k == 7) ? 1.0f : 0.0f;  // self-loop attr
    deg[i] = 1.0f;                                              // self loop
}

__global__ __launch_bounds__(256)
void k_a9_scatter(const int* __restrict__ row, const float* __restrict__ eattr,
                  float* __restrict__ A9, float* __restrict__ deg, int E) {
    int e = blockIdx.x * 256 + threadIdx.x;
    if (e >= E) return;
    int r = row[e];
    const float* a = eattr + (size_t)e * 9;
    float* d = A9 + (size_t)r * 9;
#pragma unroll
    for (int k = 0; k < 9; ++k) atomicAdd(d + k, a[k]);
    atomicAdd(deg + r, 1.0f);
}

// h[i][j] = node_emb[x[i]][j]
__global__ __launch_bounds__(256)
void k_embed(const int* __restrict__ x, const float* __restrict__ emb,
             float* __restrict__ h, int n) {
    size_t idx = (size_t)blockIdx.x * 256 + threadIdx.x;
    if (idx >= (size_t)n * EMB) return;
    int i = (int)(idx >> 7), j = (int)(idx & (EMB - 1));
    h[idx] = emb[(size_t)x[i] * EMB + j];
}

// aggcat[i][0..127]   = h[i][:]                       (self-loop contribution)
// aggcat[i][128..255] = A9[i] @ edge_w[l] + deg[i]*edge_b[l]
__global__ __launch_bounds__(256)
void k_aggcat_init(const float* __restrict__ h, const float* __restrict__ A9,
                   const float* __restrict__ deg, const float* __restrict__ ew,
                   const float* __restrict__ eb, float* __restrict__ agg, int n) {
    size_t idx = (size_t)blockIdx.x * 256 + threadIdx.x;
    if (idx >= (size_t)n * C2) return;
    int i = (int)(idx >> 8), j = (int)(idx & (C2 - 1));
    if (j < EMB) {
        agg[idx] = h[(size_t)i * EMB + j];
    } else {
        int jj = j - EMB;
        float s = deg[i] * eb[jj];
        const float* a = A9 + (size_t)i * 9;
#pragma unroll
        for (int k = 0; k < 9; ++k) s += a[k] * ew[k * EMB + jj];
        agg[idx] = s;
    }
}

// One wave32 per edge: lane L carries 4 floats. float4 gather of h[col],
// 4 atomic f32 adds into aggcat[row][0..127]. Working set is L2-resident.
__global__ __launch_bounds__(256)
void k_spmm(const int* __restrict__ row, const int* __restrict__ col,
            const float* __restrict__ h, float* __restrict__ agg, int E) {
    size_t tid = (size_t)blockIdx.x * 256 + threadIdx.x;
    int e = (int)(tid >> 5);
    if (e >= E) return;
    int lane = (int)(tid & 31);
    int r = row[e], c = col[e];
    float4 v = ((const float4*)(h + (size_t)c * EMB))[lane];
    float* dst = agg + (size_t)r * C2 + lane * 4;
    atomicAdd(dst + 0, v.x);
    atomicAdd(dst + 1, v.y);
    atomicAdd(dst + 2, v.z);
    atomicAdd(dst + 3, v.w);
}

// W[K][Ncol] f32  ->  Wt[Ncol][K] f16   (so GEMM B loads are contiguous b128)
__global__ __launch_bounds__(256)
void k_transpose_cvt(const float* __restrict__ W, _Float16* __restrict__ Wt,
                     int K, int Ncol) {
    int idx = blockIdx.x * 256 + threadIdx.x;
    if (idx >= K * Ncol) return;
    int k = idx / Ncol, nn = idx % Ncol;
    Wt[(size_t)nn * K + k] = (_Float16)W[idx];
}

// ---------------------------------------------------------------------------
// WMMA GEMM:  C[M][Ncols] = A[M][256] * Bt[Ncols][256]^T + bias  (opt. ReLU)
// Wave computes one 16-row tile x NT 16-col tiles. K loop fully unrolled.
// AF32: A is fp32 and converted to f16 in-register (v_cvt co-executes with
// the XDL pipe), eliminating a separate 77MB/layer conversion pass.
// A/B lane layout per CDNA5 ISA 7.12.2 (16-bit A 16x32):
//   lanes 0-15 : elems 0..7 = K kb..kb+7, elems 8..15 = K kb+16..kb+23 (kb=ks)
//   lanes 16-31: same with kb = ks+8
// C/D: vgpr r -> M = r + 8*(lane>=16), N = lane&15.
// ---------------------------------------------------------------------------
template <int NT, bool AF32>
__global__ __launch_bounds__(128)
void k_gemm_wmma(const void* __restrict__ Aptr, const _Float16* __restrict__ Bt,
                 const float* __restrict__ bias, float* __restrict__ Cmat,
                 int Mvalid, int Ncols, int relu) {
    const int lane  = threadIdx.x & 31;
    const int wave  = threadIdx.x >> 5;
    const int m     = blockIdx.x * 16 + (lane & 15);
    const int kboff = (lane & 16) ? 8 : 0;
    const int n0    = wave * (NT * 16);

    v8f zero = {};
    v8f acc[NT];
#pragma unroll
    for (int t = 0; t < NT; ++t) acc[t] = zero;

#pragma unroll
    for (int ks = 0; ks < KDIM; ks += 32) {
        const int kb = ks + kboff;
        v16h a;
        if constexpr (AF32) {
            const float* arow = (const float*)Aptr + (size_t)m * KDIM;
            float4 f0 = *(const float4*)(arow + kb);
            float4 f1 = *(const float4*)(arow + kb + 4);
            float4 f2 = *(const float4*)(arow + kb + 16);
            float4 f3 = *(const float4*)(arow + kb + 20);
            a[0]  = (_Float16)f0.x; a[1]  = (_Float16)f0.y;
            a[2]  = (_Float16)f0.z; a[3]  = (_Float16)f0.w;
            a[4]  = (_Float16)f1.x; a[5]  = (_Float16)f1.y;
            a[6]  = (_Float16)f1.z; a[7]  = (_Float16)f1.w;
            a[8]  = (_Float16)f2.x; a[9]  = (_Float16)f2.y;
            a[10] = (_Float16)f2.z; a[11] = (_Float16)f2.w;
            a[12] = (_Float16)f3.x; a[13] = (_Float16)f3.y;
            a[14] = (_Float16)f3.z; a[15] = (_Float16)f3.w;
        } else {
            const _Float16* arow = (const _Float16*)Aptr + (size_t)m * KDIM;
            v8h alo = *(const v8h*)(arow + kb);
            v8h ahi = *(const v8h*)(arow + kb + 16);
#pragma unroll
            for (int i = 0; i < 8; ++i) { a[i] = alo[i]; a[i + 8] = ahi[i]; }
        }
        // Hoist all NT B-fragments first: independent loads the scheduler can
        // overlap with the previous k-step's WMMA chain.
        v16h b[NT];
#pragma unroll
        for (int t = 0; t < NT; ++t) {
            const _Float16* brow = Bt + (size_t)(n0 + t * 16 + (lane & 15)) * KDIM;
            v8h blo = *(const v8h*)(brow + kb);
            v8h bhi = *(const v8h*)(brow + kb + 16);
#pragma unroll
            for (int i = 0; i < 8; ++i) { b[t][i] = blo[i]; b[t][i + 8] = bhi[i]; }
        }
#pragma unroll
        for (int t = 0; t < NT; ++t) {
            acc[t] = __builtin_amdgcn_wmma_f32_16x16x32_f16(
                false, a, false, b[t], (short)0, acc[t], false, false);
        }
    }

    const int mrow_base = blockIdx.x * 16 + ((lane & 16) ? 8 : 0);
#pragma unroll
    for (int t = 0; t < NT; ++t) {
        const int n  = n0 + t * 16 + (lane & 15);
        const float bv = bias[n];
#pragma unroll
        for (int r = 0; r < 8; ++r) {
            const int mrow = mrow_base + r;
            if (mrow < Mvalid) {
                float v = acc[t][r] + bv;
                if (relu) v = fmaxf(v, 0.0f);
                Cmat[(size_t)mrow * Ncols + n] = v;
            }
        }
    }
}

// ----------------------- BatchNorm statistics ------------------------------
__global__ __launch_bounds__(256)
void k_bn_zero(float* __restrict__ sum, float* __restrict__ sumsq) {
    sum[threadIdx.x] = 0.0f;
    sumsq[threadIdx.x] = 0.0f;
}

// Coalesced row sweep: thread t owns column t; each block reads whole 1KB rows.
// Partial per-column sums -> 262k global f32 atomics total.
__global__ __launch_bounds__(256)
void k_bn_partial(const float* __restrict__ z, int n,
                  float* __restrict__ sum, float* __restrict__ sumsq) {
    const int c = threadIdx.x;
    float s = 0.f, q = 0.f;
    for (int i = blockIdx.x; i < n; i += gridDim.x) {
        float v = z[(size_t)i * C2 + c];
        s += v; q += v * v;
    }
    atomicAdd(&sum[c], s);
    atomicAdd(&sumsq[c], q);
}

__global__ __launch_bounds__(256)
void k_bn_finalize(const float* __restrict__ sum, const float* __restrict__ sumsq,
                   int n, float* __restrict__ mean, float* __restrict__ rstd) {
    const int c = threadIdx.x;
    float mu  = sum[c] / (float)n;
    float var = sumsq[c] / (float)n - mu * mu;   // population variance (ddof=0)
    mean[c] = mu;
    rstd[c] = rsqrtf(var + BN_EPS);
}

// normalize * gamma + beta, ReLU, convert to f16 for the second WMMA GEMM
__global__ __launch_bounds__(256)
void k_bn_apply(const float* __restrict__ z, const float* __restrict__ mean,
                const float* __restrict__ rstd, const float* __restrict__ gamma,
                const float* __restrict__ beta, _Float16* __restrict__ zh, size_t count) {
    size_t idx = (size_t)blockIdx.x * 256 + threadIdx.x;
    if (idx >= count) return;
    int j = (int)(idx & (C2 - 1));
    float v = (z[idx] - mean[j]) * rstd[j] * gamma[j] + beta[j];
    v = fmaxf(v, 0.0f);
    zh[idx] = (_Float16)v;
}

// ---------------------------------------------------------------------------
extern "C" void kernel_launch(void* const* d_in, const int* in_sizes, int n_in,
                              void* d_out, int out_size, void* d_ws, size_t ws_size,
                              hipStream_t stream) {
    const int n = in_sizes[0];
    const int E = in_sizes[2] / 9;

    const int*   x        = (const int*)d_in[0];
    const int*   ei       = (const int*)d_in[1];      // row = ei[0..E), col = ei[E..2E)
    const float* eattr    = (const float*)d_in[2];
    const float* node_emb = (const float*)d_in[3];
    const float* edge_w   = (const float*)d_in[4];    // [5][9][128]
    const float* edge_b   = (const float*)d_in[5];    // [5][128]
    const float* w1       = (const float*)d_in[6];    // [5][256][256]
    const float* b1       = (const float*)d_in[7];    // [5][256]
    const float* gamma    = (const float*)d_in[8];    // [5][256]
    const float* beta     = (const float*)d_in[9];    // [5][256]
    const float* w2       = (const float*)d_in[10];   // [5][256][128]
    const float* b2       = (const float*)d_in[11];   // [5][128]
    float* out = (float*)d_out;

    char* p = (char*)d_ws;
    auto alloc = [&](size_t bytes) -> void* {
        void* r = (void*)p;
        p += (bytes + 255) & ~(size_t)255;
        return r;
    };
    float*    h      = (float*)   alloc((size_t)n * EMB * 4);   // node features
    float*    aggcat = (float*)   alloc((size_t)n * C2  * 4);   // [agg(h) | agg(edge)]
    float*    z      = (float*)   alloc((size_t)n * C2  * 4);
    _Float16* zh     = (_Float16*)alloc((size_t)n * C2  * 2);
    float*    A9     = (float*)   alloc((size_t)n * 9   * 4);
    float*    deg    = (float*)   alloc((size_t)n * 4);
    _Float16* w1t    = (_Float16*)alloc((size_t)C2 * C2 * 2);
    _Float16* w2t    = (_Float16*)alloc((size_t)EMB * C2 * 2);
    float*    bnsum  = (float*)   alloc(C2 * 4);
    float*    bnsq   = (float*)   alloc(C2 * 4);
    float*    mean   = (float*)   alloc(C2 * 4);
    float*    rstd   = (float*)   alloc(C2 * 4);
    (void)ws_size; (void)n_in; (void)out_size;

    const unsigned gN    = (unsigned)((n + 255) / 256);
    const unsigned gE    = (unsigned)((E + 255) / 256);
    const unsigned gNE   = (unsigned)(((size_t)n * EMB + 255) / 256);
    const unsigned gNC   = (unsigned)(((size_t)n * C2 + 255) / 256);
    const unsigned gSp   = (unsigned)(((size_t)E * 32 + 255) / 256);
    const unsigned Mtile = (unsigned)((n + 15) / 16);           // 3125 for n=50000

    // Layer-invariant sparse precompute + input embedding
    k_a9_init   <<<gN,  256, 0, stream>>>(A9, deg, n);
    k_a9_scatter<<<gE,  256, 0, stream>>>(ei, eattr, A9, deg, E);
    k_embed     <<<gNE, 256, 0, stream>>>(x, node_emb, h, n);

    for (int l = 0; l < NLAYER; ++l) {
        k_transpose_cvt<<<(C2 * C2 + 255) / 256, 256, 0, stream>>>(
            w1 + (size_t)l * C2 * C2, w1t, C2, C2);
        k_transpose_cvt<<<(C2 * EMB + 255) / 256, 256, 0, stream>>>(
            w2 + (size_t)l * C2 * EMB, w2t, C2, EMB);

        k_aggcat_init<<<gNC, 256, 0, stream>>>(
            h, A9, deg, edge_w + (size_t)l * 9 * EMB, edge_b + (size_t)l * EMB, aggcat, n);
        k_spmm<<<gSp, 256, 0, stream>>>(ei, ei + E, h, aggcat, E);

        // z = aggcat @ w1[l] + b1[l]   (WMMA, fp32 A converted in-register)
        k_gemm_wmma<4, true><<<Mtile, 128, 0, stream>>>(
            (const void*)aggcat, w1t, b1 + (size_t)l * C2, z, n, C2, 0);

        k_bn_zero    <<<1, 256, 0, stream>>>(bnsum, bnsq);
        k_bn_partial <<<BN_BLOCKS, 256, 0, stream>>>(z, n, bnsum, bnsq);
        k_bn_finalize<<<1, 256, 0, stream>>>(bnsum, bnsq, n, mean, rstd);
        k_bn_apply   <<<gNC, 256, 0, stream>>>(
            z, mean, rstd, gamma + (size_t)l * C2, beta + (size_t)l * C2, zh, (size_t)n * C2);

        // h = relu(z @ w2[l] + b2[l])  (WMMA, f16 A); last layer -> d_out, no ReLU
        float* dst = (l == NLAYER - 1) ? out : h;
        k_gemm_wmma<2, false><<<Mtile, 128, 0, stream>>>(
            (const void*)zh, w2t, b2 + (size_t)l * EMB, dst, n, EMB, (l < NLAYER - 1) ? 1 : 0);
    }
}
